// Encoder_inter_31971736551505
// MI455X (gfx1250) — matrix operand order repaired and checked by reading, not compiled, not run on hardware
//
#include <hip/hip_runtime.h>
#include <hip/hip_bf16.h>

typedef _Float16 v16h __attribute__((ext_vector_type(16)));
typedef _Float16 v8h  __attribute__((ext_vector_type(8)));
typedef float    v8f  __attribute__((ext_vector_type(8)));

#define B_ 32
#define N_ 512
#define T_ 24
#define D_ 64
#define H_ 256
#define M_TOTAL (B_ * N_ * T_)   // 393216

// db4-like filters (L=6) from the reference
__constant__ float c_dec_lo[6] = {-0.01565572813546454f, -0.0727326195128539f, 0.38486484686420286f,
                                  0.8525720202122554f, 0.3378976624578092f, -0.0727326195128539f};
__constant__ float c_dec_hi[6] = {0.0727326195128539f, 0.3378976624578092f, -0.8525720202122554f,
                                  0.38486484686420286f, 0.0727326195128539f, -0.01565572813546454f};
__constant__ float c_rec_lo[6] = {-0.0727326195128539f, 0.3378976624578092f, 0.8525720202122554f,
                                  0.38486484686420286f, -0.0727326195128539f, -0.01565572813546454f};
__constant__ float c_rec_hi[6] = {-0.01565572813546454f, 0.0727326195128539f, 0.38486484686420286f,
                                  -0.8525720202122554f, 0.3378976624578092f, 0.0727326195128539f};

// ------------------------------------------------------------------
// Kernel 0: convert fp32 weights to f16 in workspace
// ------------------------------------------------------------------
__global__ void cvt_weights_kernel(const float* __restrict__ W1, const float* __restrict__ W2,
                                   _Float16* __restrict__ w1h, _Float16* __restrict__ w2h) {
    int i = blockIdx.x * blockDim.x + threadIdx.x;
    if (i < H_ * D_) w1h[i] = (_Float16)W1[i];
    if (i < H_ * H_) w2h[i] = (_Float16)W2[i];
}

// ------------------------------------------------------------------
// Kernel 1: h0 = disentangle_inter(x) (wavelet along N axis), f32 -> f16
// block = (b, t, 16-wide d chunk); LDS-stage the 512-long rows.
// ------------------------------------------------------------------
__global__ void __launch_bounds__(256) wavelet_kernel(const float* __restrict__ x,
                                                      _Float16* __restrict__ h0) {
    __shared__ float xs[16 * N_];   // 32 KB, xs[d][n]
    const int blk = blockIdx.x;
    const int dc = blk & 3;                 // d chunk 0..3
    const int t  = (blk >> 2) % T_;
    const int b  = blk / (4 * T_);
    const int d0 = dc * 16;

    const float* xbase = x + ((size_t)b * N_ * T_ + t) * D_ + d0;

    // load: xs[d * N_ + n] = x[b][n][t][d0+d]
    for (int i = threadIdx.x; i < 16 * N_; i += 256) {
        int n = i >> 4;
        int d = i & 15;
        xs[d * N_ + n] = xbase[(size_t)n * (T_ * D_) + d];
    }
    __syncthreads();

    // y[n] = sum_j rec_lo[j]*cA[m] + rec_hi[j]*(2*cD[m]),  m = (n+4-j)/2 when even
    for (int i = threadIdx.x; i < 16 * N_; i += 256) {
        int d = i >> 9;            // i / 512
        int n = i & (N_ - 1);
        const float* row = xs + d * N_;
        float y = 0.f;
#pragma unroll
        for (int j = 0; j < 6; ++j) {
            int q = n + 4 - j;     // index into zero-upsampled array
            if (q >= 0 && q <= 514 && ((q & 1) == 0)) {
                int m = q >> 1;    // 0..257
                float ca = 0.f, cd = 0.f;
#pragma unroll
                for (int jj = 0; jj < 6; ++jj) {
                    int e = 6 - jj + 2 * m;   // symmetric-extension index, 1..520
                    int xi = (e < 5) ? (4 - e) : ((e <= 516) ? (e - 5) : (1028 - e));
                    float xv = row[xi];
                    ca = fmaf(c_dec_lo[jj], xv, ca);
                    cd = fmaf(c_dec_hi[jj], xv, cd);
                }
                y = fmaf(c_rec_lo[j], ca, y);
                y = fmaf(c_rec_hi[j], 2.0f * cd, y);
            }
        }
        size_t rowi = ((size_t)b * N_ + n) * T_ + t;   // GEMM M index
        h0[rowi * D_ + d0 + d] = (_Float16)y;
    }
}

// ------------------------------------------------------------------
// WMMA fragment loaders (wave32 layouts from CDNA5 ISA 7.12.2)
// ------------------------------------------------------------------
// A (16x32 f16): lane<16 -> M=lane, halves K[0..7],K[16..23]; lane>=16 -> K[8..15],K[24..31]
__device__ __forceinline__ v16h load_a_frag(const _Float16* p) {
    v8h lo = *(const v8h*)(p);
    v8h hi = *(const v8h*)(p + 16);
    v16h a;
#pragma unroll
    for (int i = 0; i < 8; ++i) { a[i] = lo[i]; a[8 + i] = hi[i]; }
    return a;
}
// B (32x16 f16) from row-major W[N][K]: lane<16 -> N=lane, K[0..7],K[16..23]; lane>=16 -> K[8..15],K[24..31]
__device__ __forceinline__ v16h load_b_frag(const _Float16* p) {
    v8h lo = *(const v8h*)(p);
    v8h hi = *(const v8h*)(p + 16);
    v16h b;
#pragma unroll
    for (int i = 0; i < 8; ++i) { b[i] = lo[i]; b[8 + i] = hi[i]; }
    return b;
}

// ------------------------------------------------------------------
// Kernel 2: fused MLP.  block = 64 M rows, 8 waves.
//   stage1: h1(64x256 f16, LDS) = relu(h0 @ W1^T + b1)   (K=64 -> 2 wmma/tile)
//   stage2: out = relu(h1 @ W2^T + b2)                   (K=256 -> 8 wmma/tile)
// ------------------------------------------------------------------
#define LDH 264   // 256 + 8 halves padding -> conflict-free ds_load_b128

__global__ void __launch_bounds__(256) mlp_kernel(const _Float16* __restrict__ h0,
                                                  const _Float16* __restrict__ w1h,
                                                  const _Float16* __restrict__ w2h,
                                                  const float* __restrict__ b1,
                                                  const float* __restrict__ b2,
                                                  float* __restrict__ out) {
    __shared__ _Float16 h1[64 * LDH];   // 33.8 KB

    const int tid   = threadIdx.x;
    const int lane  = tid & 31;
    const int wave  = tid >> 5;
    const int nloc  = lane & 15;         // N index within tile (B & C/D layouts)
    const int hiH   = lane >> 4;         // 0 or 1
    const int kbA   = hiH ? 8 : 0;       // A per-lane K base
    const int kbB   = hiH ? 8 : 0;       // B per-lane K base
    const int m_hi  = hiH * 8;           // C/D per-lane M base
    const size_t m0 = (size_t)blockIdx.x * 64;

    const int mt  = wave >> 1;           // 0..3 : 16-row M tile
    const int nt0 = (wave & 1) * 8;      // 0 or 8: first of 8 N tiles

    // ---------------- stage 1 ----------------
    const _Float16* aRow = h0 + (m0 + (size_t)(mt * 16 + (lane & 15))) * D_;
    v16h a0 = load_a_frag(aRow + 0 + kbA);       // K 0..31
    v16h a1 = load_a_frag(aRow + 32 + kbA);      // K 32..63

#pragma unroll
    for (int ni = 0; ni < 8; ++ni) {
        const int nt  = nt0 + ni;
        const int col = nt * 16 + nloc;          // global h index
        const _Float16* wRow = w1h + col * D_ + kbB;
        v16h bf0 = load_b_frag(wRow);
        v16h bf1 = load_b_frag(wRow + 32);
        v8f acc = {};
        acc = __builtin_amdgcn_wmma_f32_16x16x32_f16(false, a0, false, bf0, (short)0, acc, false, false);
        acc = __builtin_amdgcn_wmma_f32_16x16x32_f16(false, a1, false, bf1, (short)0, acc, false, false);
        const float bias = b1[col];
#pragma unroll
        for (int r = 0; r < 8; ++r) {
            float v = fmaxf(acc[r] + bias, 0.f);
            h1[(mt * 16 + m_hi + r) * LDH + col] = (_Float16)v;
        }
    }
    __syncthreads();

    // ---------------- stage 2 ----------------
    v16h a2[8];
    const _Float16* aL = h1 + (mt * 16 + (lane & 15)) * LDH + kbA;
#pragma unroll
    for (int kc = 0; kc < 8; ++kc) a2[kc] = load_a_frag(aL + kc * 32);

#pragma unroll
    for (int ni = 0; ni < 8; ++ni) {
        const int nt  = nt0 + ni;
        const int col = nt * 16 + nloc;          // global g index
        const _Float16* wRow = w2h + col * H_ + kbB;
        v8f acc = {};
#pragma unroll
        for (int kc = 0; kc < 8; ++kc) {
            v16h bf = load_b_frag(wRow + kc * 32);
            acc = __builtin_amdgcn_wmma_f32_16x16x32_f16(false, a2[kc], false, bf, (short)0, acc, false, false);
        }
        const float bias = b2[col];
        const size_t orow = m0 + mt * 16 + m_hi;
#pragma unroll
        for (int r = 0; r < 8; ++r) {
            float v = fmaxf(acc[r] + bias, 0.f);
            out[(orow + r) * (size_t)H_ + col] = v;
        }
    }
}

// ------------------------------------------------------------------
extern "C" void kernel_launch(void* const* d_in, const int* in_sizes, int n_in,
                              void* d_out, int out_size, void* d_ws, size_t ws_size,
                              hipStream_t stream) {
    const float* x  = (const float*)d_in[0];
    const float* W1 = (const float*)d_in[1];
    const float* b1 = (const float*)d_in[2];
    const float* W2 = (const float*)d_in[3];
    const float* b2 = (const float*)d_in[4];
    float* out = (float*)d_out;

    char* ws = (char*)d_ws;
    _Float16* w1h = (_Float16*)ws;                         // 32 KB
    _Float16* w2h = (_Float16*)(ws + 32768);               // 128 KB
    _Float16* h0  = (_Float16*)(ws + 32768 + 131072);      // 48 MB (M_TOTAL x 64 f16)

    cvt_weights_kernel<<<(H_ * H_ + 255) / 256, 256, 0, stream>>>(W1, W2, w1h, w2h);
    wavelet_kernel<<<B_ * T_ * 4, 256, 0, stream>>>(x, h0);
    mlp_kernel<<<M_TOTAL / 64, 256, 0, stream>>>(h0, w1h, w2h, b1, b2, out);
}
